// SwitchFeedForward_86131274154913
// MI455X (gfx1250) — matrix operand order repaired and checked by reading, not compiled, not run on hardware
//
#include <hip/hip_runtime.h>
#include <hip/hip_bf16.h>

typedef __attribute__((ext_vector_type(16))) __bf16 v16bf;
typedef __attribute__((ext_vector_type(8)))  float  v8f;

#define N_TOK 8192
#define DDIM  2048
#define ODIM  2048
#define NEXP  8
#define BM    128
#define BN    128
#define BK    32
#define LDA   40   // LDS row stride in bf16 elems (80 B, 16-B aligned)
#define LDB   40

// ---------------------------------------------------------------- init ------
__global__ void init_kernel(float* __restrict__ out_counts,
                            float* __restrict__ out_psum,
                            int* __restrict__ counters) {
    int t = threadIdx.x;
    if (t < NEXP) {
        out_counts[t] = 0.0f;
        out_psum[t]   = 0.0f;
        counters[t]   = 0;
    }
}

// -------------------------------------------------------------- router -----
__global__ __launch_bounds__(256)
void router_kernel(const float* __restrict__ x,
                   const float* __restrict__ sw,     // [D, E]
                   const float* __restrict__ sb,     // [E]
                   float* __restrict__ out_counts,   // [E]
                   float* __restrict__ out_psum,     // [E]
                   float* __restrict__ out_pmax,     // [N]
                   int*   __restrict__ counters,     // [E]
                   int*   __restrict__ token_list) { // [E][N]
    const int wave = threadIdx.x >> 5;
    const int lane = threadIdx.x & 31;
    const int n    = blockIdx.x * 8 + wave;

    float acc[NEXP];
#pragma unroll
    for (int e = 0; e < NEXP; ++e) acc[e] = 0.0f;

    const float* xrow = x + (size_t)n * DDIM;
    for (int d = lane; d < DDIM; d += 32) {
        float xv = xrow[d];
        const float4* w4 = (const float4*)(sw + (size_t)d * NEXP);
        float4 w0 = w4[0], w1 = w4[1];
        acc[0] += xv * w0.x; acc[1] += xv * w0.y;
        acc[2] += xv * w0.z; acc[3] += xv * w0.w;
        acc[4] += xv * w1.x; acc[5] += xv * w1.y;
        acc[6] += xv * w1.z; acc[7] += xv * w1.w;
    }
#pragma unroll
    for (int e = 0; e < NEXP; ++e) {
        float v = acc[e];
#pragma unroll
        for (int off = 16; off > 0; off >>= 1)
            v += __shfl_xor(v, off, 32);
        acc[e] = v + sb[e];
    }
    float mx = acc[0];
#pragma unroll
    for (int e = 1; e < NEXP; ++e) mx = fmaxf(mx, acc[e]);
    float p[NEXP]; float ssum = 0.0f;
#pragma unroll
    for (int e = 0; e < NEXP; ++e) { p[e] = __expf(acc[e] - mx); ssum += p[e]; }
    float inv = 1.0f / ssum;
    int   best = 0; float pbv = p[0];
#pragma unroll
    for (int e = 1; e < NEXP; ++e) if (p[e] > pbv) { pbv = p[e]; best = e; }

    if (lane == 0) {
        out_pmax[n] = pbv * inv;
#pragma unroll
        for (int e = 0; e < NEXP; ++e) atomicAdd(out_psum + e, p[e] * inv);
        atomicAdd(out_counts + best, 1.0f);
        int slot = atomicAdd(counters + best, 1);
        token_list[best * N_TOK + slot] = n;
    }
}

// --------------------------------------------------- grouped expert GEMM ---
// grid = (ODIM/BN, N_TOK/BM, NEXP); blocks beyond expert's token count exit.
// 8 waves, each owning a 32x64 strip of the 128x128 C tile:
//   2 A frags x 4 B frags -> 8 v_wmma_f32_16x16x32_bf16 per K step.
__global__ __launch_bounds__(256)
void moe_gemm_kernel(const float* __restrict__ x,          // [N, D]
                     const float* __restrict__ ew,         // [E, D, O]
                     const float* __restrict__ eb,         // [E, O]
                     const float* __restrict__ pmax,       // [N]
                     const int*   __restrict__ counters,   // [E]
                     const int*   __restrict__ token_list, // [E][N]
                     float*       __restrict__ out) {      // [N, O]
    const int e   = blockIdx.z;
    const int cnt = counters[e];
    const int m0  = blockIdx.y * BM;
    if (m0 >= cnt) return;                 // block-uniform: whole waves exit
    const int rows = min(BM, cnt - m0);
    const int n0   = blockIdx.x * BN;

    __shared__ __align__(16) __bf16 lds_a[BM * LDA];  // [m][k]
    __shared__ __align__(16) __bf16 lds_b[BN * LDB];  // [n][k] (transposed)
    __shared__ int   toks[BM];
    __shared__ float scal[BM];

    const int tid = threadIdx.x;
    {
        int idx0 = m0 + tid;
        int idx1 = m0 + tid + 128;
        int tk0  = (idx0 < cnt) ? token_list[e * N_TOK + idx0] : -1;
        int tk1  = (idx1 < cnt) ? token_list[e * N_TOK + idx1] : -1;
        toks[tid]       = tk0;
        toks[tid + 128] = tk1;
        scal[tid]       = (tk0 >= 0) ? pmax[tk0] : 0.0f;
        scal[tid + 128] = (tk1 >= 0) ? pmax[tk1] : 0.0f;
    }
    __syncthreads();

    const int lane = tid & 31;
    const int wave = tid >> 5;
    const int wm   = wave & 3;   // 0..3 -> 32-row strip
    const int wn   = wave >> 2;  // 0..1 -> 64-col strip

    v8f c[2][4];
#pragma unroll
    for (int a = 0; a < 2; ++a)
#pragma unroll
        for (int j = 0; j < 4; ++j) c[a][j] = (v8f){};

    // ---- staging coordinates (hoisted out of K loop)
    const int ar  = tid >> 1;            // A: row 0..127
    const int ak  = (tid & 1) * 16;      // A: k offset {0,16}
    const int bcg = (tid & 31) * 4;      // B: col group start {0,4,...,124}
    const int bkt = (tid >> 5) * 4;      // B: k offset {0,4,...,28}

    // Branchless gather: invalid rows read token 0; epilogue masks them.
    int tka = toks[ar]; if (tka < 0) tka = 0;
    const float4* aptr =
        (const float4*)(x + (size_t)tka * DDIM + ak);            // + kb/4
    const size_t  eoff = (size_t)e * DDIM * ODIM;
    const float*  bptr = ew + eoff + (size_t)bkt * ODIM + n0 + bcg; // +(kb+i)*O

    // ---- software pipeline: prefetch tile kb=0 into registers
    float4 pa[4];
    float4 pb[4];   // pb[i] = 4 consecutive cols at k = kb + bkt + i
#pragma unroll
    for (int q = 0; q < 4; ++q) pa[q] = aptr[q];
#pragma unroll
    for (int i = 0; i < 4; ++i)
        pb[i] = *(const float4*)(bptr + (size_t)i * ODIM);

    for (int kb = 0; kb < DDIM; kb += BK) {
        // ---- A: convert 16 f32 -> bf16, store 2x b128
        {
            union { __bf16 h[16]; uint4 q[2]; } u;
#pragma unroll
            for (int q = 0; q < 4; ++q) {
                u.h[q * 4 + 0] = (__bf16)pa[q].x;
                u.h[q * 4 + 1] = (__bf16)pa[q].y;
                u.h[q * 4 + 2] = (__bf16)pa[q].z;
                u.h[q * 4 + 3] = (__bf16)pa[q].w;
            }
            *(uint4*)&lds_a[ar * LDA + ak]     = u.q[0];
            *(uint4*)&lds_a[ar * LDA + ak + 8] = u.q[1];
        }
        // ---- B: 4x4 register transpose -> 4x ds_store_b64 into [n][k]
        {
            union { __bf16 h[4]; uint2 q; } u0, u1, u2, u3;
            u0.h[0] = (__bf16)pb[0].x; u0.h[1] = (__bf16)pb[1].x;
            u0.h[2] = (__bf16)pb[2].x; u0.h[3] = (__bf16)pb[3].x;
            u1.h[0] = (__bf16)pb[0].y; u1.h[1] = (__bf16)pb[1].y;
            u1.h[2] = (__bf16)pb[2].y; u1.h[3] = (__bf16)pb[3].y;
            u2.h[0] = (__bf16)pb[0].z; u2.h[1] = (__bf16)pb[1].z;
            u2.h[2] = (__bf16)pb[2].z; u2.h[3] = (__bf16)pb[3].z;
            u3.h[0] = (__bf16)pb[0].w; u3.h[1] = (__bf16)pb[1].w;
            u3.h[2] = (__bf16)pb[2].w; u3.h[3] = (__bf16)pb[3].w;
            *(uint2*)&lds_b[(bcg + 0) * LDB + bkt] = u0.q;
            *(uint2*)&lds_b[(bcg + 1) * LDB + bkt] = u1.q;
            *(uint2*)&lds_b[(bcg + 2) * LDB + bkt] = u2.q;
            *(uint2*)&lds_b[(bcg + 3) * LDB + bkt] = u3.q;
        }
        __syncthreads();

        // ---- issue next tile's global loads (overlap with WMMA below)
        const int kn = kb + BK;
        if (kn < DDIM) {
            const float4* an = aptr + (kn >> 2);
            const float*  bn = bptr + (size_t)kn * ODIM;
#pragma unroll
            for (int q = 0; q < 4; ++q) pa[q] = an[q];
#pragma unroll
            for (int i = 0; i < 4; ++i)
                pb[i] = *(const float4*)(bn + (size_t)i * ODIM);
        }

        // ---- fragments per ISA layout
        // A lane: row = lane%16, K = {k0..k0+7} U {k0+16..k0+23}, k0=(lane/16)*8
        const int ak0 = (lane >> 4) * 8;
        union { v16bf v; uint4 q[2]; } af[2];
#pragma unroll
        for (int a = 0; a < 2; ++a) {
            const int arow = wm * 32 + a * 16 + (lane & 15);
            af[a].q[0] = *(const uint4*)&lds_a[arow * LDA + ak0];
            af[a].q[1] = *(const uint4*)&lds_a[arow * LDA + ak0 + 16];
        }
        // B lane: col = lane%16, K = {k0..k0+7} U {k0+16..k0+23}, k0=(lane/16)*8
        const int bk0 = (lane >> 4) * 8;
        union { v16bf v; uint4 q[2]; } bf[4];
#pragma unroll
        for (int j = 0; j < 4; ++j) {
            const int bcol = wn * 64 + j * 16 + (lane & 15);
            bf[j].q[0] = *(const uint4*)&lds_b[bcol * LDB + bk0];
            bf[j].q[1] = *(const uint4*)&lds_b[bcol * LDB + bk0 + 16];
        }

#pragma unroll
        for (int a = 0; a < 2; ++a)
#pragma unroll
            for (int j = 0; j < 4; ++j)
                c[a][j] = __builtin_amdgcn_wmma_f32_16x16x32_bf16(
                              false, af[a].v, false, bf[j].v,
                              (short)0, c[a][j], false, false);

        __syncthreads();
    }

    // ---- epilogue: bias + ReLU + scale by route_prob_max, scatter by token
#pragma unroll
    for (int a = 0; a < 2; ++a) {
        const int mbase = wm * 32 + a * 16 + (lane >> 4) * 8;
#pragma unroll
        for (int j = 0; j < 4; ++j) {
            const int col  = n0 + wn * 64 + j * 16 + (lane & 15);
            const float bias = eb[(size_t)e * ODIM + col];
#pragma unroll
            for (int i = 0; i < 8; ++i) {
                int mloc = mbase + i;
                if (mloc < rows) {
                    int   tk = toks[mloc];
                    float s  = scal[mloc];
                    float v  = c[a][j][i] + bias;
                    v = v > 0.0f ? v : 0.0f;
                    out[(size_t)tk * ODIM + col] = v * s;
                }
            }
        }
    }
}

// -------------------------------------------------------------- launch -----
extern "C" void kernel_launch(void* const* d_in, const int* in_sizes, int n_in,
                              void* d_out, int out_size, void* d_ws, size_t ws_size,
                              hipStream_t stream) {
    (void)in_sizes; (void)n_in; (void)out_size; (void)ws_size;
    const float* x  = (const float*)d_in[0];
    const float* sw = (const float*)d_in[1];
    const float* sb = (const float*)d_in[2];
    const float* ew = (const float*)d_in[3];
    const float* eb = (const float*)d_in[4];

    float* out        = (float*)d_out;
    float* out_counts = out + (size_t)N_TOK * ODIM;      // [E]
    float* out_psum   = out_counts + NEXP;               // [E]
    float* out_pmax   = out_psum + NEXP;                 // [N]

    int* counters   = (int*)d_ws;                        // [E]
    int* token_list = (int*)((char*)d_ws + 1024);        // [E][N]

    init_kernel<<<1, 32, 0, stream>>>(out_counts, out_psum, counters);

    router_kernel<<<N_TOK / 8, 256, 0, stream>>>(
        x, sw, sb, out_counts, out_psum, out_pmax, counters, token_list);

    dim3 grid(ODIM / BN, N_TOK / BM, NEXP);
    moe_gemm_kernel<<<grid, 256, 0, stream>>>(
        x, ew, eb, out_pmax, counters, token_list, out);
}
